// SelfAttention_49443663512179
// MI455X (gfx1250) — matrix operand order repaired and checked by reading, not compiled, not run on hardware
//
#include <hip/hip_runtime.h>

// ---------------------------------------------------------------------------
// Self-attention layer for MI455X (gfx1250), bf16 WMMA pipeline.
//   x:[2,2048,1024] f32 -> QKV gemm -> flash attention -> out gemm -> f32
// ---------------------------------------------------------------------------

typedef __attribute__((ext_vector_type(16))) __bf16 v16bf;
typedef __attribute__((ext_vector_type(8)))  float  v8f;
typedef unsigned short ushort_t;

union Frag16 { v16bf v; uint4 q[2]; };

__device__ __forceinline__ ushort_t f2bf(float f) {
    union { float f; unsigned u; } c; c.f = f;
    unsigned lsb = (c.u >> 16) & 1u;
    c.u += 0x7fffu + lsb;                 // round-to-nearest-even
    return (ushort_t)(c.u >> 16);
}

// ---------------------------------------------------------------------------
// Conversion / packing kernels
// ---------------------------------------------------------------------------
__global__ void cvt_f32_bf16(const float* __restrict__ in,
                             ushort_t* __restrict__ out, int n) {
    int i = blockIdx.x * 256 + threadIdx.x;
    if (i < n) out[i] = f2bf(in[i]);
}

// in: [K][N] f32 row-major  ->  out: [N][K] bf16 row-major  (weight transpose)
__global__ void cvt_f32_bf16_T(const float* __restrict__ in,
                               ushort_t* __restrict__ out, int K, int N) {
    int i = blockIdx.x * 256 + threadIdx.x;
    if (i < K * N) {
        int n = i / K;
        int k = i - n * K;
        out[i] = f2bf(in[(size_t)k * N + n]);
    }
}

// ---------------------------------------------------------------------------
// bf16 GEMM: C[M,N] = A[M,K] * Bt[N,K]^T (+bias)
// mode 0: f32 output to Cout
// mode 1: QKV scatter epilogue -> Qb [BH,S,64], Kb [BH,S,64], Vt [BH,64,S]
// Block tile 128x128x32, 8 waves, wave tile 32x64 (2x4 WMMA frags).
// ---------------------------------------------------------------------------
#define BM 128
#define BN 128
#define BK 32
#define LDT 40   // BK + 8 pad (80B row stride, 16B aligned)

__global__ __launch_bounds__(256)
void gemm_bf16(const ushort_t* __restrict__ A, const ushort_t* __restrict__ Bt,
               const float* __restrict__ bias, float* __restrict__ Cout,
               ushort_t* __restrict__ Qb, ushort_t* __restrict__ Kb,
               ushort_t* __restrict__ Vt, int M, int N, int K, int mode) {
    __shared__ __align__(16) ushort_t As[BM][LDT];
    __shared__ __align__(16) ushort_t Bs[BN][LDT];

    const int m0 = blockIdx.x * BM;
    const int n0 = blockIdx.y * BN;
    const int tid  = threadIdx.x;
    const int lane = tid & 31;
    const int wid  = tid >> 5;
    const int wm   = wid & 3;      // 4 waves along M (32 rows each)
    const int wn   = wid >> 2;     // 2 waves along N (64 cols each)
    const int lr   = lane & 15;
    const int hf   = lane >> 4;

    // cooperative staging assignment: 2 threads per row, 16 elems (32B) each
    const int srow = tid >> 1;
    const int shalf = tid & 1;
    const ushort_t* aptr = A  + (size_t)(m0 + srow) * K + shalf * 16;
    const ushort_t* bptr = Bt + (size_t)(n0 + srow) * K + shalf * 16;

    v8f acc[2][4];
#pragma unroll
    for (int mf = 0; mf < 2; ++mf)
#pragma unroll
        for (int nf = 0; nf < 4; ++nf) acc[mf][nf] = (v8f){};

    const int ksteps = K / BK;
    for (int kt = 0; kt < ksteps; ++kt) {
        __syncthreads();
        {
            const uint4* as = (const uint4*)(aptr + kt * BK);
            const uint4* bs = (const uint4*)(bptr + kt * BK);
            *(uint4*)(&As[srow][shalf * 16])     = as[0];
            *(uint4*)(&As[srow][shalf * 16 + 8]) = as[1];
            *(uint4*)(&Bs[srow][shalf * 16])     = bs[0];
            *(uint4*)(&Bs[srow][shalf * 16 + 8]) = bs[1];
            if (kt + 1 < ksteps) {            // -> global_prefetch_b8 (near-cache)
                __builtin_prefetch(aptr + (kt + 1) * BK, 0, 3);
                __builtin_prefetch(bptr + (kt + 1) * BK, 0, 3);
            }
        }
        __syncthreads();

        Frag16 af[2], bf[4];
#pragma unroll
        for (int mf = 0; mf < 2; ++mf) {
            // A layout: lane<16 holds K 0..7 & 16..23, lane>=16 holds 8..15 & 24..31
            const ushort_t* p = &As[wm * 32 + mf * 16 + lr][hf * 8];
            af[mf].q[0] = *(const uint4*)(p);
            af[mf].q[1] = *(const uint4*)(p + 16);
        }
#pragma unroll
        for (int nf = 0; nf < 4; ++nf) {
            // B layout: lane<16 holds K 0..7 & 16..23, lane>=16 holds 8..15 & 24..31
            const ushort_t* p = &Bs[wn * 64 + nf * 16 + lr][hf * 8];
            bf[nf].q[0] = *(const uint4*)(p);
            bf[nf].q[1] = *(const uint4*)(p + 16);
        }
#pragma unroll
        for (int mf = 0; mf < 2; ++mf)
#pragma unroll
            for (int nf = 0; nf < 4; ++nf)
                acc[mf][nf] = __builtin_amdgcn_wmma_f32_16x16x32_bf16(
                    false, af[mf].v, false, bf[nf].v,
                    (short)0, acc[mf][nf], false, false);
    }

    // epilogue: C layout -> lane<16 VGPR r = row r, lane>=16 VGPR r = row r+8
#pragma unroll
    for (int mf = 0; mf < 2; ++mf) {
#pragma unroll
        for (int nf = 0; nf < 4; ++nf) {
            const int col = n0 + wn * 64 + nf * 16 + lr;
            const float bv = bias[col];
#pragma unroll
            for (int r = 0; r < 8; ++r) {
                const int row = m0 + wm * 32 + mf * 16 + hf * 8 + r;
                const float v = acc[mf][nf][r] + bv;
                if (mode == 0) {
                    Cout[(size_t)row * N + col] = v;
                } else {
                    // qkv split: col -> head h, which (q/k/v), dim d
                    const int h = col / 192;
                    const int rr = col - h * 192;
                    const int which = rr >> 6;
                    const int d = rr & 63;
                    const int b = row >> 11;        // /2048
                    const int s = row & 2047;
                    const int bh = b * 16 + h;
                    const ushort_t bw = f2bf(v);
                    if (which == 0)      Qb[((size_t)bh * 2048 + s) * 64 + d] = bw;
                    else if (which == 1) Kb[((size_t)bh * 2048 + s) * 64 + d] = bw;
                    else                 Vt[((size_t)bh * 64 + d) * 2048 + s] = bw;
                }
            }
        }
    }
}

// ---------------------------------------------------------------------------
// Flash attention: grid (B*H, S/128). 8 waves/block, each wave owns 16 q-rows.
// KV staged with double-buffered GLOBAL_LOAD_ASYNC_TO_LDS_B128 (ASYNCcnt),
// online softmax, P re-layout via wave-private LDS.
// ---------------------------------------------------------------------------
#define NKV 32   // 2048 / 64 kv tiles

__global__ __launch_bounds__(256)
void flash_attn(const ushort_t* __restrict__ Q, const ushort_t* __restrict__ Kb,
                const ushort_t* __restrict__ Vt, ushort_t* __restrict__ AttnOut) {
    __shared__ __align__(16) ushort_t Ks[2][64][72];  // [buf][kv_tok][d]
    __shared__ __align__(16) ushort_t Vs[2][64][72];  // [buf][d][kv_tok]
    __shared__ __align__(16) ushort_t Ps[8][16][72];  // per-wave P tile

    const int bh   = blockIdx.x;          // 0..31
    const int q0   = blockIdx.y * 128;
    const int tid  = threadIdx.x;
    const int lane = tid & 31;
    const int w    = tid >> 5;
    const int lr   = lane & 15;
    const int hf   = lane >> 4;
    const float scale = 0.125f;           // 1/sqrt(64)

    // Q fragments (held for whole kernel): A layout, k-steps d=0..31, 32..63
    Frag16 qa[2];
    {
        const ushort_t* qb = Q + ((size_t)bh * 2048 + q0 + w * 16 + lr) * 64;
#pragma unroll
        for (int ks = 0; ks < 2; ++ks) {
            const int c0 = ks * 32 + hf * 8;
            qa[ks].q[0] = *(const uint4*)(qb + c0);
            qa[ks].q[1] = *(const uint4*)(qb + c0 + 16);
        }
    }

    float mrow[8], lrow[8];
    v8f o[4];
#pragma unroll
    for (int r = 0; r < 8; ++r) { mrow[r] = -3.0e38f; lrow[r] = 0.0f; }
#pragma unroll
    for (int nf = 0; nf < 4; ++nf) o[nf] = (v8f){};

    const int srow = tid >> 2;            // 64 rows, 4 chunks of 16 elems
    const int sch  = (tid & 3) * 16;
    const ushort_t* gKbase = Kb + ((size_t)bh * 2048 + srow) * 64 + sch;
    const ushort_t* gVbase = Vt + ((size_t)bh * 64 + srow) * 2048 + sch;

    // issue async loads for kv tile 0 into buffer 0
    {
        unsigned lK = (unsigned)(uintptr_t)&Ks[0][srow][sch];
        unsigned lV = (unsigned)(uintptr_t)&Vs[0][srow][sch];
        asm volatile("global_load_async_to_lds_b128 %0, %1, off"
                     :: "v"(lK), "v"(gKbase) : "memory");
        asm volatile("global_load_async_to_lds_b128 %0, %1, off"
                     :: "v"(lV), "v"(gVbase) : "memory");
    }

    for (int it = 0; it < NKV; ++it) {
        const int buf = it & 1;
        // wait for this wave's async writes, then publish tile across workgroup
        asm volatile("s_wait_asynccnt 0x0" ::: "memory");
        __syncthreads();

        // prefetch next kv tile into the other buffer (reads of it finished
        // before the barrier above, so overwrite is safe)
        if (it + 1 < NKV) {
            const int nxt = (it + 1) * 64;
            unsigned lK = (unsigned)(uintptr_t)&Ks[buf ^ 1][srow][sch];
            unsigned lV = (unsigned)(uintptr_t)&Vs[buf ^ 1][srow][sch];
            const ushort_t* gK = gKbase + (size_t)nxt * 64; // +nxt tokens
            const ushort_t* gV = gVbase + nxt;              // +nxt cols
            asm volatile("global_load_async_to_lds_b128 %0, %1, off"
                         :: "v"(lK), "v"(gK) : "memory");
            asm volatile("global_load_async_to_lds_b128 %0, %1, off"
                         :: "v"(lV), "v"(gV) : "memory");
        }

        // S = Q K^T : 4 N-frags (kv tokens) x 2 k-steps (head dim)
        v8f sc[4];
#pragma unroll
        for (int nf = 0; nf < 4; ++nf) sc[nf] = (v8f){};
#pragma unroll
        for (int nf = 0; nf < 4; ++nf) {
#pragma unroll
            for (int ks = 0; ks < 2; ++ks) {
                Frag16 b;
                const ushort_t* p = &Ks[buf][nf * 16 + lr][ks * 32 + hf * 8];
                b.q[0] = *(const uint4*)(p);
                b.q[1] = *(const uint4*)(p + 16);
                sc[nf] = __builtin_amdgcn_wmma_f32_16x16x32_bf16(
                    false, qa[ks].v, false, b.v, (short)0, sc[nf], false, false);
            }
        }

        // online softmax (rows grouped per half-wave, butterfly over 16 lanes)
#pragma unroll
        for (int nf = 0; nf < 4; ++nf)
#pragma unroll
            for (int r = 0; r < 8; ++r) sc[nf][r] *= scale;

        float mx[8], al[8], rs[8];
#pragma unroll
        for (int r = 0; r < 8; ++r)
            mx[r] = fmaxf(fmaxf(sc[0][r], sc[1][r]), fmaxf(sc[2][r], sc[3][r]));
#pragma unroll
        for (int off = 8; off >= 1; off >>= 1)
#pragma unroll
            for (int r = 0; r < 8; ++r)
                mx[r] = fmaxf(mx[r], __shfl_xor(mx[r], off, 16));
#pragma unroll
        for (int r = 0; r < 8; ++r) {
            const float mn = fmaxf(mrow[r], mx[r]);
            al[r] = __expf(mrow[r] - mn);
            mrow[r] = mn;
        }
#pragma unroll
        for (int nf = 0; nf < 4; ++nf)
#pragma unroll
            for (int r = 0; r < 8; ++r)
                sc[nf][r] = __expf(sc[nf][r] - mrow[r]);
#pragma unroll
        for (int r = 0; r < 8; ++r)
            rs[r] = (sc[0][r] + sc[1][r]) + (sc[2][r] + sc[3][r]);
#pragma unroll
        for (int off = 8; off >= 1; off >>= 1)
#pragma unroll
            for (int r = 0; r < 8; ++r)
                rs[r] += __shfl_xor(rs[r], off, 16);
#pragma unroll
        for (int r = 0; r < 8; ++r) lrow[r] = lrow[r] * al[r] + rs[r];
#pragma unroll
        for (int nf = 0; nf < 4; ++nf)
#pragma unroll
            for (int r = 0; r < 8; ++r) o[nf][r] *= al[r];

        // P: C-layout -> bf16 A-layout via wave-private LDS tile
#pragma unroll
        for (int nf = 0; nf < 4; ++nf)
#pragma unroll
            for (int r = 0; r < 8; ++r)
                Ps[w][hf * 8 + r][nf * 16 + lr] = f2bf(sc[nf][r]);

        // O += P V : 4 N-frags (head dim) x 2 k-steps (kv tokens)
#pragma unroll
        for (int nf = 0; nf < 4; ++nf) {
#pragma unroll
            for (int ks = 0; ks < 2; ++ks) {
                Frag16 a, b;
                const ushort_t* pp = &Ps[w][lr][ks * 32 + hf * 8];
                a.q[0] = *(const uint4*)(pp);
                a.q[1] = *(const uint4*)(pp + 16);
                const ushort_t* vp = &Vs[buf][nf * 16 + lr][ks * 32 + hf * 8];
                b.q[0] = *(const uint4*)(vp);
                b.q[1] = *(const uint4*)(vp + 16);
                o[nf] = __builtin_amdgcn_wmma_f32_16x16x32_bf16(
                    false, a.v, false, b.v, (short)0, o[nf], false, false);
            }
        }
    }

    // normalize and store to AttnOut [B,S,H*64] bf16
    const int bidx = bh >> 4, h = bh & 15;
    float inv[8];
#pragma unroll
    for (int r = 0; r < 8; ++r) inv[r] = 1.0f / lrow[r];
    const size_t rowbase = (size_t)bidx * 2048 + q0 + w * 16;
#pragma unroll
    for (int nf = 0; nf < 4; ++nf)
#pragma unroll
        for (int r = 0; r < 8; ++r)
            AttnOut[(rowbase + hf * 8 + r) * 1024 + h * 64 + nf * 16 + lr] =
                f2bf(o[nf][r] * inv[r]);
}

// ---------------------------------------------------------------------------
// Host-side launch
// ---------------------------------------------------------------------------
extern "C" void kernel_launch(void* const* d_in, const int* in_sizes, int n_in,
                              void* d_out, int out_size, void* d_ws, size_t ws_size,
                              hipStream_t stream) {
    const float* x     = (const float*)d_in[0];   // [2,2048,1024]
    const float* W_qkv = (const float*)d_in[1];   // [1024,3072]
    const float* b_qkv = (const float*)d_in[2];   // [3072]
    const float* W_out = (const float*)d_in[3];   // [1024,1024]
    const float* b_out = (const float*)d_in[4];   // [1024]
    float* out = (float*)d_out;                   // [2,2048,1024]

    const int M = 4096, D = 1024;
    ushort_t* X16     = (ushort_t*)d_ws;               // 4096*1024
    ushort_t* WqkvT   = X16   + (size_t)M * D;         // 3072*1024
    ushort_t* WoutT   = WqkvT + (size_t)3 * D * D;     // 1024*1024
    ushort_t* Qb      = WoutT + (size_t)D * D;         // 32*2048*64
    ushort_t* Kb      = Qb    + (size_t)32 * 2048 * 64;
    ushort_t* Vt      = Kb    + (size_t)32 * 2048 * 64;
    ushort_t* AttnOut = Vt    + (size_t)32 * 2048 * 64; // 4096*1024

    cvt_f32_bf16<<<(M * D) / 256, 256, 0, stream>>>(x, X16, M * D);
    cvt_f32_bf16_T<<<(3 * D * D) / 256, 256, 0, stream>>>(W_qkv, WqkvT, D, 3 * D);
    cvt_f32_bf16_T<<<(D * D) / 256, 256, 0, stream>>>(W_out, WoutT, D, D);

    // QKV projection with scatter epilogue
    gemm_bf16<<<dim3(M / BM, (3 * D) / BN), 256, 0, stream>>>(
        X16, WqkvT, b_qkv, nullptr, Qb, Kb, Vt, M, 3 * D, D, 1);

    // flash attention over 32 (b,h) pairs x 16 q-blocks
    flash_attn<<<dim3(32, 2048 / 128), 256, 0, stream>>>(Qb, Kb, Vt, AttnOut);

    // output projection (fp32 + bias)
    gemm_bf16<<<dim3(M / BM, D / BN), 256, 0, stream>>>(
        AttnOut, WoutT, b_out, out, nullptr, nullptr, nullptr, M, D, D, 0);
}
